// CompressedMoeExperts_67954972557526
// MI455X (gfx1250) — matrix operand order repaired and checked by reading, not compiled, not run on hardware
//
#include <hip/hip_runtime.h>
#include <hip/hip_bf16.h>
#include <math.h>

// ---------------------------------------------------------------------------
// MoE experts for MI455X (gfx1250, wave32, WMMA).
// fp32 -> bf16 convert, V_WMMA_F32_16X16X32_BF16, fp32 accumulate.
// bf16 tiles staged with GLOBAL_LOAD_ASYNC_TO_LDS_B128 (ASYNCcnt path),
// fp32 weight tiles staged via VGPR + v_cvt_pk_bf16_f32 + ds_store.
// Double-buffered LDS, one barrier per K-step.
// ---------------------------------------------------------------------------

typedef __attribute__((ext_vector_type(16))) __bf16 v16bf;
typedef __attribute__((ext_vector_type(8)))  __bf16 v8bf;
typedef __attribute__((ext_vector_type(8)))  float  v8f;
typedef __attribute__((ext_vector_type(4)))  float  v4f;

constexpr int T  = 2048;
constexpr int D  = 1024;
constexpr int FF = 1536;
constexpr int E  = 8;
constexpr int K  = 2;

// LDS row stride in elements for 32-deep K tiles. 48*2B = 96B rows keep every
// 16B chunk aligned for ds_load_b128 / async-to-LDS b128.
constexpr int LDT = 48;

#define WMMA_BF16(a, b, c) \
    __builtin_amdgcn_wmma_f32_16x16x32_bf16(false, (a), false, (b), (short)0, (c), false, false)

__device__ inline void wait_asynccnt0() {
#if __has_builtin(__builtin_amdgcn_s_wait_asynccnt)
    __builtin_amdgcn_s_wait_asynccnt(0);
#else
    asm volatile("s_wait_asynccnt 0x0" ::: "memory");
#endif
}

// 32 bytes global(bf16) -> LDS via the CDNA5 async DMA path (ASYNCcnt).
// INST_OFFSET applies to both LDS and memory addresses (ISA 10.x async ops).
__device__ inline void stage_async32(const __bf16* g, __bf16* l) {
    unsigned lds = (unsigned)(uintptr_t)l;  // flat LDS aperture: low 32 bits = LDS addr
    asm volatile("global_load_async_to_lds_b128 %0, %1, off\n\t"
                 "global_load_async_to_lds_b128 %0, %1, off offset:16"
                 :: "v"(lds), "v"(g) : "memory");
}

// --- WMMA fragment loaders (ISA 7.12.2 layouts) ---------------------------
// A (16x32 bf16, MxK): lanes 0-15 -> M=lane,   K {0..7,16..23};
//                      lanes 16-31 -> M=lane&15, K {8..15,24..31}.
__device__ inline v16bf frag_a(const __bf16* tile, int row0, int lane) {
    const int r  = row0 + (lane & 15);
    const int lo = (lane < 16) ? 0 : 8;
    v8bf x = *(const v8bf*)(tile + r * LDT + lo);
    v8bf y = *(const v8bf*)(tile + r * LDT + lo + 16);
    return __builtin_shufflevector(x, y, 0, 1, 2, 3, 4, 5, 6, 7,
                                         8, 9, 10, 11, 12, 13, 14, 15);
}

// B (32x16 bf16, KxN) stored in LDS as tile[col][k]:
// lanes 0-15: col=lane, K {0..7,16..23}; lanes 16-31: col=lane&15, K {8..15,24..31}.
__device__ inline v16bf frag_b(const __bf16* tile, int col0, int lane) {
    const int c  = col0 + (lane & 15);
    const int ko = (lane < 16) ? 0 : 8;
    v8bf x = *(const v8bf*)(tile + c * LDT + ko);
    v8bf y = *(const v8bf*)(tile + c * LDT + ko + 16);
    return __builtin_shufflevector(x, y, 0, 1, 2, 3, 4, 5, 6, 7,
                                         8, 9, 10, 11, 12, 13, 14, 15);
}

__device__ inline v8bf cvt8_bf16(v4f a, v4f b) {
    v8bf o;
    o[0] = (__bf16)a[0]; o[1] = (__bf16)a[1]; o[2] = (__bf16)a[2]; o[3] = (__bf16)a[3];
    o[4] = (__bf16)b[0]; o[5] = (__bf16)b[1]; o[6] = (__bf16)b[2]; o[7] = (__bf16)b[3];
    return o;
}

// Load 8 fp32 from global, convert, store 16B to LDS (emits v_cvt_pk_bf16_f32).
__device__ inline void stage_w8(const float* gp, __bf16* lp) {
    v4f a = *(const v4f*)gp;
    v4f b = *(const v4f*)(gp + 4);
    *(v8bf*)lp = cvt8_bf16(a, b);
}

// --- Kernel 1: combine[e,t] from top-k routing (thread owns token t) ------
__global__ __launch_bounds__(256) void moe_combine_k(const int* __restrict__ idx,
                                                     const float* __restrict__ w,
                                                     float* __restrict__ combine) {
    int t = blockIdx.x * 256 + threadIdx.x;
    if (t >= T) return;
    int   i0 = idx[t * K + 0], i1 = idx[t * K + 1];
    float w0 = w[t * K + 0],   w1 = w[t * K + 1];
#pragma unroll
    for (int e = 0; e < E; ++e)
        combine[e * T + t] = (i0 == e ? w0 : 0.0f) + (i1 == e ? w1 : 0.0f);
}

// --- Kernel 2: hidden_states fp32 -> bf16 ---------------------------------
__global__ __launch_bounds__(256) void moe_cvt_x_k(const float* __restrict__ X,
                                                   __bf16* __restrict__ Xbf) {
    size_t i = ((size_t)blockIdx.x * 256 + threadIdx.x) * 8;
    v4f a = *(const v4f*)(X + i);
    v4f b = *(const v4f*)(X + i + 4);
    *(v8bf*)(Xbf + i) = cvt8_bf16(a, b);
}

// --- Kernel 3: gate+up GEMMs, SiLU, combine-scale, store H' bf16 ----------
// Block tile 128(T) x 64(FF); 8 waves as 4(M) x 2(N), wave tile 32x32.
__global__ __launch_bounds__(256) void moe_gateup_k(const __bf16* __restrict__ Xbf,
                                                    const float* __restrict__ Wg,
                                                    const float* __restrict__ Wu,
                                                    const float* __restrict__ combine,
                                                    __bf16* __restrict__ Hs) {
    __shared__ alignas(16) __bf16 sX[2][128 * LDT];
    __shared__ alignas(16) __bf16 sG[2][64 * LDT];
    __shared__ alignas(16) __bf16 sU[2][64 * LDT];

    const int e    = blockIdx.z;
    const int t0   = blockIdx.y * 128;
    const int f0   = blockIdx.x * 64;
    const int tid  = threadIdx.x;
    const int lane = tid & 31;
    const int wave = tid >> 5;
    const int wm   = wave & 3;   // 4 waves over M (32 rows each)
    const int wn   = wave >> 2;  // 2 waves over N (32 cols each)

    const __bf16* xg = Xbf + (size_t)t0 * D;
    const float*  wg = Wg + ((size_t)e * FF + f0) * D;
    const float*  wu = Wu + ((size_t)e * FF + f0) * D;

    // Per-thread staging coordinates.
    const int xrow = tid >> 1, xoff = (tid & 1) * 16;  // X tile: 128 x 32, 32B/thread
    const int wrow = tid >> 2, woff = (tid & 3) * 8;   // W tiles: 64 x 32, 8 f32/thread

    v8f aG[2][2] = {};
    v8f aU[2][2] = {};

    auto stage = [&](int k0, int buf) {
        stage_async32(xg + (size_t)xrow * D + k0 + xoff, &sX[buf][xrow * LDT + xoff]);
        stage_w8(wg + (size_t)wrow * D + k0 + woff, &sG[buf][wrow * LDT + woff]);
        stage_w8(wu + (size_t)wrow * D + k0 + woff, &sU[buf][wrow * LDT + woff]);
    };

    constexpr int NK = D / 32;
    stage(0, 0);

    for (int kt = 0; kt < NK; ++kt) {
        const int cur = kt & 1;
        wait_asynccnt0();     // tile kt async stores complete (kt+1 not yet issued)
        __syncthreads();      // all staging (incl. ds_store path) visible
        if (kt + 1 < NK) stage((kt + 1) * 32, cur ^ 1);

        const __bf16* bX = sX[cur];
        const __bf16* bG = sG[cur];
        const __bf16* bU = sU[cur];

        v16bf a0  = frag_a(bX, wm * 32,      lane);
        v16bf a1  = frag_a(bX, wm * 32 + 16, lane);
        v16bf bg0 = frag_b(bG, wn * 32,      lane);
        v16bf bg1 = frag_b(bG, wn * 32 + 16, lane);
        v16bf bu0 = frag_b(bU, wn * 32,      lane);
        v16bf bu1 = frag_b(bU, wn * 32 + 16, lane);

        aG[0][0] = WMMA_BF16(a0, bg0, aG[0][0]);
        aG[0][1] = WMMA_BF16(a0, bg1, aG[0][1]);
        aG[1][0] = WMMA_BF16(a1, bg0, aG[1][0]);
        aG[1][1] = WMMA_BF16(a1, bg1, aG[1][1]);
        aU[0][0] = WMMA_BF16(a0, bu0, aU[0][0]);
        aU[0][1] = WMMA_BF16(a0, bu1, aU[0][1]);
        aU[1][0] = WMMA_BF16(a1, bu0, aU[1][0]);
        aU[1][1] = WMMA_BF16(a1, bu1, aU[1][1]);
    }

    // Epilogue: h = silu(g) * u * combine[e,t]; store H' bf16.
    // C/D layout: VGPR r, lanes 0-15 -> M=r; lanes 16-31 -> M=r+8 (N=lane&15).
    const int mhi = (lane >> 4) * 8;
#pragma unroll
    for (int mi = 0; mi < 2; ++mi) {
#pragma unroll
        for (int ni = 0; ni < 2; ++ni) {
            int fcol = f0 + wn * 32 + ni * 16 + (lane & 15);
#pragma unroll
            for (int r = 0; r < 8; ++r) {
                int   t = t0 + wm * 32 + mi * 16 + mhi + r;
                float g = aG[mi][ni][r];
                float u = aU[mi][ni][r];
                float h = (g / (1.0f + __expf(-g))) * u * combine[e * T + t];
                Hs[((size_t)e * T + t) * FF + fcol] = (__bf16)h;
            }
        }
    }
}

// --- Kernel 4: out[t,d] = sum_e sum_f H'[e,t,f] * Wd[e,d,f] ---------------
// Block tile 128(T) x 128(D); 8 waves as 2(M) x 4(N), wave tile 64x32.
// K dimension flattened over (expert, FF) for seamless double buffering.
__global__ __launch_bounds__(256) void moe_down_k(const __bf16* __restrict__ Hs,
                                                  const float* __restrict__ Wd,
                                                  float* __restrict__ out) {
    __shared__ alignas(16) __bf16 sA[2][128 * LDT];
    __shared__ alignas(16) __bf16 sB[2][128 * LDT];

    const int t0   = blockIdx.y * 128;
    const int d0   = blockIdx.x * 128;
    const int tid  = threadIdx.x;
    const int lane = tid & 31;
    const int wave = tid >> 5;
    const int wm   = wave & 1;   // 2 waves over M (64 rows each)
    const int wn   = wave >> 1;  // 4 waves over N (32 cols each)

    const int arow = tid >> 1, aoff = (tid & 1) * 16;  // both tiles: 128 x 32

    v8f acc[4][2] = {};

    constexpr int KS = FF / 32;       // K-steps per expert
    constexpr int NS = E * KS;        // total K-steps

    auto stage = [&](int s, int buf) {
        const int se = s / KS;
        const int k0 = (s - se * KS) * 32;
        const __bf16* hA = Hs + ((size_t)se * T + t0) * FF + (size_t)arow * FF + k0 + aoff;
        const float*  wd = Wd + ((size_t)se * D + d0) * FF + (size_t)arow * FF + k0 + aoff;
        stage_async32(hA, &sA[buf][arow * LDT + aoff]);
        stage_w8(wd,     &sB[buf][arow * LDT + aoff]);
        stage_w8(wd + 8, &sB[buf][arow * LDT + aoff + 8]);
    };

    stage(0, 0);

    for (int s = 0; s < NS; ++s) {
        const int cur = s & 1;
        wait_asynccnt0();
        __syncthreads();
        if (s + 1 < NS) stage(s + 1, cur ^ 1);

        const __bf16* bA = sA[cur];
        const __bf16* bB = sB[cur];

        v16bf b0 = frag_b(bB, wn * 32,      lane);
        v16bf b1 = frag_b(bB, wn * 32 + 16, lane);
#pragma unroll
        for (int mi = 0; mi < 4; ++mi) {
            v16bf a = frag_a(bA, wm * 64 + mi * 16, lane);
            acc[mi][0] = WMMA_BF16(a, b0, acc[mi][0]);
            acc[mi][1] = WMMA_BF16(a, b1, acc[mi][1]);
        }
    }

    const int mhi = (lane >> 4) * 8;
#pragma unroll
    for (int mi = 0; mi < 4; ++mi) {
#pragma unroll
        for (int ni = 0; ni < 2; ++ni) {
            int d = d0 + wn * 32 + ni * 16 + (lane & 15);
#pragma unroll
            for (int r = 0; r < 8; ++r) {
                int t = t0 + wm * 64 + mi * 16 + mhi + r;
                out[(size_t)t * D + d] = acc[mi][ni][r];
            }
        }
    }
}

// ---------------------------------------------------------------------------
extern "C" void kernel_launch(void* const* d_in, const int* in_sizes, int n_in,
                              void* d_out, int out_size, void* d_ws, size_t ws_size,
                              hipStream_t stream) {
    const float* X   = (const float*)d_in[0];  // [T, D]
    const int*   tki = (const int*)d_in[1];    // [T, K]
    const float* tkw = (const float*)d_in[2];  // [T, K]
    const float* Wg  = (const float*)d_in[3];  // [E, FF, D]
    const float* Wu  = (const float*)d_in[4];  // [E, FF, D]
    const float* Wd  = (const float*)d_in[5];  // [E, D, FF]
    float*       out = (float*)d_out;          // [T, D]

    char* ws = (char*)d_ws;
    __bf16* Xbf     = (__bf16*)ws;                                   // T*D*2    = 4 MB
    float*  combine = (float*)(ws + (size_t)T * D * 2);              // E*T*4    = 64 KB
    __bf16* Hs      = (__bf16*)(ws + (size_t)T * D * 2 +
                                     (size_t)E * T * 4);             // E*T*FF*2 = 48 MB

    moe_combine_k<<<dim3(T / 256), dim3(256), 0, stream>>>(tki, tkw, combine);
    moe_cvt_x_k<<<dim3((T * D) / (8 * 256)), dim3(256), 0, stream>>>(X, Xbf);
    moe_gateup_k<<<dim3(FF / 64, T / 128, E), dim3(256), 0, stream>>>(Xbf, Wg, Wu, combine, Hs);
    moe_down_k<<<dim3(D / 128, T / 128), dim3(256), 0, stream>>>(Hs, Wd, out);
}
